// AdaptiveMoEWithRL_4355096838529
// MI455X (gfx1250) — matrix operand order repaired and checked by reading, not compiled, not run on hardware
//
#include <hip/hip_runtime.h>
#include <hip/hip_bf16.h>

typedef __attribute__((ext_vector_type(16))) _Float16 v16h;
typedef __attribute__((ext_vector_type(8)))  _Float16 v8h;
typedef __attribute__((ext_vector_type(8)))  float    v8f;
typedef __attribute__((ext_vector_type(4)))  float    v4f;

#define NEXP   10
#define INSZ   3072
#define HID    256
#define OUTSZ  10
#define BATCH  16384
#define MTILE  64            // rows per workgroup
#define KP     128           // K panel width (elements)
#define XS_STRIDE 136        // padded LDS stride (halves) for x panel
#define HS_STRIDE 264        // padded LDS stride (halves) for h tile

// ---- fragment loaders (wave32 WMMA, contiguous k order: k = 16*(lane/16) + s) ----

// B fragment 32x16 f16: lane holds column n = lane%16, K = k0 + (lane/16)*16 .. +15,
// packed 2 halves per VGPR ascending -> 16 contiguous halves in memory (N-major weights).
static __device__ __forceinline__ v16h load_frag16(const _Float16* p) {
  v8h p0 = *(const v8h*)(p);
  v8h p1 = *(const v8h*)(p + 8);
  v16h r;
#pragma unroll
  for (int i = 0; i < 8; ++i) { r[i] = p0[i]; r[i + 8] = p1[i]; }
  return r;
}

// A fragment 16x32 f16: lane holds row m = lane%16; halves 0..15 = K off+0..15,
// where off = (lane/16)*16. Two 16B chunks from a row.
static __device__ __forceinline__ v16h load_a_frag(const _Float16* row, int kh, int hsel) {
  v8h p0 = *(const v8h*)(row + kh + hsel * 16);
  v8h p1 = *(const v8h*)(row + kh + hsel * 16 + 8);
  v16h r;
#pragma unroll
  for (int i = 0; i < 8; ++i) { r[i] = p0[i]; r[i + 8] = p1[i]; }
  return r;
}

// ---- prep: W1 [E][K=3072][N=256] f32 -> W1T [E][N=256][K=3072] f16 (tiled transpose) ----
__global__ __launch_bounds__(256) void k_transpose_w1(const float* __restrict__ W1,
                                                      _Float16* __restrict__ w1t) {
  __shared__ _Float16 t[32][33];
  const int b  = blockIdx.x;
  const int e  = b / ((INSZ / 32) * (HID / 32));
  const int r  = b % ((INSZ / 32) * (HID / 32));
  const int k0 = (r / (HID / 32)) * 32;
  const int n0 = (r % (HID / 32)) * 32;
  const float* src = W1 + ((size_t)e * INSZ + k0) * HID + n0;
#pragma unroll
  for (int i = 0; i < 4; ++i) {
    const int lin = threadIdx.x + 256 * i;
    const int kk = lin >> 5, nn = lin & 31;
    t[kk][nn] = (_Float16)src[(size_t)kk * HID + nn];
  }
  __syncthreads();
  _Float16* dst = w1t + ((size_t)e * HID + n0) * INSZ + k0;
#pragma unroll
  for (int i = 0; i < 4; ++i) {
    const int lin = threadIdx.x + 256 * i;
    const int nn = lin >> 5, kk = lin & 31;
    dst[(size_t)nn * INSZ + kk] = t[kk][nn];
  }
}

// ---- prep: gate_w [K=3072][10] -> gwt [16][3072] f16 (zero-padded rows),
//            W2 [E][H=256][10]   -> w2t [E][16][256] f16 (zero-padded cols) ----
__global__ __launch_bounds__(256) void k_prep_small(const float* __restrict__ gate_w,
                                                    const float* __restrict__ W2,
                                                    _Float16* __restrict__ gwt,
                                                    _Float16* __restrict__ w2t) {
  const int idx = blockIdx.x * 256 + threadIdx.x;
  if (idx < 16 * INSZ) {
    const int n = idx / INSZ, k = idx % INSZ;
    gwt[idx] = (_Float16)((n < NEXP) ? gate_w[(size_t)k * NEXP + n] : 0.0f);
  }
  if (idx < NEXP * 16 * HID) {
    const int e = idx / (16 * HID);
    const int rem = idx % (16 * HID);
    const int o = rem / HID, h = rem % HID;
    w2t[idx] = (_Float16)((o < OUTSZ) ? W2[((size_t)e * HID + h) * OUTSZ + o] : 0.0f);
  }
}

// ---- fused dense-MoE kernel ----
__global__ __launch_bounds__(256) void k_moe_fused(
    const float*    __restrict__ x,
    const float*    __restrict__ b1,
    const float*    __restrict__ b2,
    const float*    __restrict__ gate_b,
    const _Float16* __restrict__ w1t,
    const _Float16* __restrict__ gwt,
    const _Float16* __restrict__ w2t,
    float*          __restrict__ out)
{
  __shared__ _Float16 xs[MTILE * XS_STRIDE];  // x panel, f16
  __shared__ _Float16 hs[MTILE * HS_STRIDE];  // h tile, f16
  __shared__ float    gl[MTILE * 16];         // gate logits -> softmax probs

  const int tid  = threadIdx.x;
  const int wid  = tid >> 5;       // 8 waves
  const int lane = tid & 31;
  const int hsel = lane >> 4;      // 0 or 1
  const int lrow = lane & 15;
  const int m0   = blockIdx.x * MTILE;

  v8f outacc, gacc;
#pragma unroll
  for (int i = 0; i < 8; ++i) { outacc[i] = 0.f; gacc[i] = 0.f; }

  for (int e = 0; e < NEXP; ++e) {
    v8f acc[4][2];
#pragma unroll
    for (int mt = 0; mt < 4; ++mt)
#pragma unroll
      for (int nt = 0; nt < 2; ++nt)
#pragma unroll
        for (int i = 0; i < 8; ++i) acc[mt][nt][i] = 0.f;

    const _Float16* w1e   = w1t + (size_t)e * HID * INSZ;
    const int       ncol0 = wid * 32;

    for (int p = 0; p < INSZ / KP; ++p) {
      const int kp = p * KP;
      __syncthreads();
      // stage x[64 x 128] f32 -> f16 LDS (cooperative, coalesced)
#pragma unroll
      for (int j = 0; j < 4; ++j) {
        const int lin  = tid + 256 * j;
        const int row  = lin >> 4;
        const int colh = (lin & 15) << 3;
        const float* sp = x + (size_t)(m0 + row) * INSZ + kp + colh;
        v4f a0 = *(const v4f*)sp;
        v4f a1 = *(const v4f*)(sp + 4);
        v8h h8;
        h8[0] = (_Float16)a0[0]; h8[1] = (_Float16)a0[1];
        h8[2] = (_Float16)a0[2]; h8[3] = (_Float16)a0[3];
        h8[4] = (_Float16)a1[0]; h8[5] = (_Float16)a1[1];
        h8[6] = (_Float16)a1[2]; h8[7] = (_Float16)a1[3];
        *(v8h*)&xs[row * XS_STRIDE + colh] = h8;
      }
      __syncthreads();
#pragma unroll
      for (int kc = 0; kc < KP / 32; ++kc) {
        const int kh = kc * 32;
        v16h A[4];
#pragma unroll
        for (int mt = 0; mt < 4; ++mt)
          A[mt] = load_a_frag(&xs[(16 * mt + lrow) * XS_STRIDE], kh, hsel);
        const int kg = kp + kh + hsel * 16;
        v16h B0 = load_frag16(w1e + (size_t)(ncol0 + lrow) * INSZ + kg);
        v16h B1 = load_frag16(w1e + (size_t)(ncol0 + 16 + lrow) * INSZ + kg);
#pragma unroll
        for (int mt = 0; mt < 4; ++mt) {
          acc[mt][0] = __builtin_amdgcn_wmma_f32_16x16x32_f16(
              false, A[mt], false, B0, (short)0, acc[mt][0], false, false);
          acc[mt][1] = __builtin_amdgcn_wmma_f32_16x16x32_f16(
              false, A[mt], false, B1, (short)0, acc[mt][1], false, false);
        }
        if (e == 0 && wid < 4) {
          // gate GEMM piggybacks on expert 0's K loop; wave w owns m-tile w.
          // Re-load the A fragment at the wave's own row (direct ds_load) instead
          // of dynamically indexing A[wid], which lowered to a long cndmask chain.
          v16h Ag = load_a_frag(&xs[(16 * wid + lrow) * XS_STRIDE], kh, hsel);
          v16h Bg = load_frag16(gwt + (size_t)lrow * INSZ + kg);
          gacc = __builtin_amdgcn_wmma_f32_16x16x32_f16(
              false, Ag, false, Bg, (short)0, gacc, false, false);
        }
      }
    }

    // ---- epilogue for expert e ----
    if (e == 0 && wid < 4) {  // gate logits + bias -> LDS
      const float gb = (lrow < NEXP) ? gate_b[lrow] : 0.f;
#pragma unroll
      for (int r = 0; r < 8; ++r)
        gl[(16 * wid + r + 8 * hsel) * 16 + lrow] = gacc[r] + gb;
    }
    __syncthreads();
    if (e == 0 && tid < MTILE) {  // softmax over 10 experts per row
      float l[NEXP]; float mx = -1e30f;
#pragma unroll
      for (int o = 0; o < NEXP; ++o) { l[o] = gl[tid * 16 + o]; mx = fmaxf(mx, l[o]); }
      float s = 0.f;
#pragma unroll
      for (int o = 0; o < NEXP; ++o) { l[o] = __expf(l[o] - mx); s += l[o]; }
      const float inv = 1.f / s;
#pragma unroll
      for (int o = 0; o < NEXP; ++o) gl[tid * 16 + o] = l[o] * inv;
    }
    // bias + relu, h (f32->f16) -> LDS, all waves
#pragma unroll
    for (int mt = 0; mt < 4; ++mt)
#pragma unroll
      for (int nt = 0; nt < 2; ++nt) {
        const int hcol = ncol0 + 16 * nt + lrow;
        const float b1v = b1[e * HID + hcol];
#pragma unroll
        for (int r = 0; r < 8; ++r) {
          float v = acc[mt][nt][r] + b1v;
          v = fmaxf(v, 0.f);
          hs[(16 * mt + r + 8 * hsel) * HS_STRIDE + hcol] = (_Float16)v;
        }
      }
    __syncthreads();
    if (wid < 4) {  // layer 2 via WMMA (O padded to 16), gate-weighted combine in regs
      v8f eo;
#pragma unroll
      for (int i = 0; i < 8; ++i) eo[i] = 0.f;
      const _Float16* w2e  = w2t + ((size_t)e * 16 + lrow) * HID;
      const _Float16* hrow = &hs[(16 * wid + lrow) * HS_STRIDE];
#pragma unroll
      for (int kc = 0; kc < HID / 32; ++kc) {
        const int kh = kc * 32;
        v16h A = load_a_frag(hrow, kh, hsel);
        v16h B = load_frag16(w2e + kh + hsel * 16);
        eo = __builtin_amdgcn_wmma_f32_16x16x32_f16(
            false, A, false, B, (short)0, eo, false, false);
      }
      const float b2v = (lrow < OUTSZ) ? b2[e * OUTSZ + lrow] : 0.f;
#pragma unroll
      for (int r = 0; r < 8; ++r) {
        const int row = 16 * wid + r + 8 * hsel;
        const float g = gl[row * 16 + e];
        outacc[r] += g * (eo[r] + b2v);
      }
    }
  }

  if (wid < 4 && lrow < OUTSZ) {
#pragma unroll
    for (int r = 0; r < 8; ++r)
      out[(size_t)(m0 + 16 * wid + r + 8 * hsel) * OUTSZ + lrow] = outacc[r];
  }
}

extern "C" void kernel_launch(void* const* d_in, const int* in_sizes, int n_in,
                              void* d_out, int out_size, void* d_ws, size_t ws_size,
                              hipStream_t stream) {
  const float* x      = (const float*)d_in[0];
  const float* W1     = (const float*)d_in[1];
  const float* b1     = (const float*)d_in[2];
  const float* W2     = (const float*)d_in[3];
  const float* b2     = (const float*)d_in[4];
  const float* gate_w = (const float*)d_in[5];
  const float* gate_b = (const float*)d_in[6];
  float* out = (float*)d_out;

  // ws layout: W1T f16 (15.0 MiB) | gwt f16 (96 KiB) | w2t f16 (80 KiB)
  const size_t w1t_bytes = (size_t)NEXP * HID * INSZ * sizeof(_Float16);
  const size_t gwt_bytes = (size_t)16 * INSZ * sizeof(_Float16);
  const size_t w2t_bytes = (size_t)NEXP * 16 * HID * sizeof(_Float16);
  if (ws_size < w1t_bytes + gwt_bytes + w2t_bytes) return;  // scratch too small

  char* ws = (char*)d_ws;
  _Float16* w1t = (_Float16*)ws;
  _Float16* gwt = (_Float16*)(ws + w1t_bytes);
  _Float16* w2t = (_Float16*)(ws + w1t_bytes + gwt_bytes);

  k_transpose_w1<<<NEXP * (INSZ / 32) * (HID / 32), 256, 0, stream>>>(W1, w1t);
  k_prep_small<<<(16 * INSZ + 255) / 256, 256, 0, stream>>>(gate_w, W2, gwt, w2t);
  k_moe_fused<<<BATCH / MTILE, 256, 0, stream>>>(x, b1, b2, gate_b, w1t, gwt, w2t, out);
}
